// TtT5Attention_75977971466511
// MI455X (gfx1250) — matrix operand order repaired and checked by reading, not compiled, not run on hardware
//
#include <hip/hip_runtime.h>
#include <hip/hip_bf16.h>
#include <math.h>

// ---------------------------------------------------------------------------
// T5 encoder self-attention for MI455X (gfx1250, wave32, WMMA bf16)
// B=4, S=1024, D=1024, H=16, DK=64, NUM_BUCKETS=32, MAX_DIST=128
// ---------------------------------------------------------------------------

typedef __bf16 bf16_t;
typedef __attribute__((ext_vector_type(8)))  __bf16 v8bf;
typedef __attribute__((ext_vector_type(16))) __bf16 v16bf;
typedef __attribute__((ext_vector_type(8)))  float  v8f;

static constexpr int B_   = 4;
static constexpr int S_   = 1024;
static constexpr int D_   = 1024;
static constexpr int H_   = 16;
static constexpr int DK_  = 64;
static constexpr int INNER = H_ * DK_;   // 1024
static constexpr int M_   = B_ * S_;     // 4096

// ---- optional CDNA5 async global->LDS path (guarded; manual fallback) -----
#if defined(__has_builtin)
#if __has_builtin(__builtin_amdgcn_global_load_async_to_lds_b128)
#define T5_HAVE_ASYNC_LDS 1
#endif
#endif

#ifdef T5_HAVE_ASYNC_LDS
// Builtin signature: (v4i addrspace(1)*, v4i addrspace(3)*, imm offset, imm cpol)
// where v4i is a GCC-style vector of 4 ints (per the hipcc diagnostic).
// Generic LDS pointers carry the LDS byte offset in their low 32 bits (flat
// aperture rule), so integer casts are value-correct.
typedef int t5_i4 __attribute__((vector_size(16)));
#define T5_TO_GM(p)  ((__attribute__((address_space(1))) t5_i4*)(unsigned long long)(p))
#define T5_TO_LDS(p) ((__attribute__((address_space(3))) t5_i4*)(unsigned int)(unsigned long long)(p))
__device__ __forceinline__ void t5_wait_async0() {
#if __has_builtin(__builtin_amdgcn_s_wait_asynccnt)
    __builtin_amdgcn_s_wait_asynccnt(0);
#else
    asm volatile("s_wait_asynccnt 0" ::: "memory");
#endif
}
#endif

// ---------------------------------------------------------------------------
// WMMA wrapper: D = A(16x32 bf16) * B(32x16 bf16) + C(16x16 f32)
// ---------------------------------------------------------------------------
__device__ __forceinline__ v8f wmma_bf16(v16bf a, v16bf b, v8f c) {
    return __builtin_amdgcn_wmma_f32_16x16x32_bf16(
        /*neg_a=*/false, a, /*neg_b=*/false, b,
        /*c_mod=*/(short)0, c, /*reuse_a=*/false, /*reuse_b=*/false);
}

// A-fragment (16x32, bf16), row-major source, row stride ld (elements).
// ISA layout: lane<16 -> row M=lane, k in {0..7, 16..23};
//             lane>=16 -> row M=lane-16, k in {8..15, 24..31}.
__device__ __forceinline__ v16bf frag_a(const bf16_t* p, int ld) {
    const int lane = threadIdx.x & 31;
    const int r    = lane & 15;
    const int kh   = (lane >> 4) << 3;           // 0 or 8
    const bf16_t* q0 = p + r * ld + kh;
    v8bf lo = *(const v8bf*)q0;
    v8bf hi = *(const v8bf*)(q0 + 16);
    v16bf f;
#pragma unroll
    for (int i = 0; i < 8; ++i) { f[i] = lo[i]; f[i + 8] = hi[i]; }
    return f;
}

// B-fragment (32x16, bf16): B[k][n]. Source stored as rows indexed by n with
// contiguous k (i.e. Bsrc[n][k]), row stride ld.
// ISA layout: lane n (<16) holds col n, k in {0..7, 16..23}; lane n+16 holds col n, k in {8..15, 24..31}.
__device__ __forceinline__ v16bf frag_b(const bf16_t* p, int ld) {
    const int lane = threadIdx.x & 31;
    const int n    = lane & 15;
    const int kh   = (lane >> 4) << 3;           // 0 or 8
    const bf16_t* q0 = p + n * ld + kh;
    v8bf lo = *(const v8bf*)q0;
    v8bf hi = *(const v8bf*)(q0 + 16);
    v16bf f;
#pragma unroll
    for (int i = 0; i < 8; ++i) { f[i] = lo[i]; f[i + 8] = hi[i]; }
    return f;
}

// pack 8 floats (two float4) into v8bf
__device__ __forceinline__ v8bf pack8(float4 f0, float4 f1) {
    v8bf p;
    p[0] = (bf16_t)f0.x; p[1] = (bf16_t)f0.y; p[2] = (bf16_t)f0.z; p[3] = (bf16_t)f0.w;
    p[4] = (bf16_t)f1.x; p[5] = (bf16_t)f1.y; p[6] = (bf16_t)f1.z; p[7] = (bf16_t)f1.w;
    return p;
}

// ---------------------------------------------------------------------------
// Kernel 0: relative-position bias LUT, head-major:
// lut2[h*2048 + (k - q + S - 1)] = rel_bias[bucket(k-q)][h]
// ---------------------------------------------------------------------------
__global__ void t5_bias_lut_kernel(const float* __restrict__ rel_bias,
                                   float* __restrict__ lut2) {
    int i = blockIdx.x * blockDim.x + threadIdx.x;
    if (i >= 2 * S_ - 1) return;
    int rp = i - (S_ - 1);                 // mem - ctx
    int rb = (rp > 0) ? 16 : 0;            // NUM_BUCKETS/2
    int n  = abs(rp);
    int bucket;
    if (n < 8) {                           // max_exact = 8
        bucket = rb + n;
    } else {
        float l = logf((float)n * (1.0f / 8.0f)) * (1.0f / logf(16.0f)) * 8.0f;
        int large = 8 + (int)l;
        if (large > 15) large = 15;
        bucket = rb + large;
    }
#pragma unroll
    for (int h = 0; h < H_; ++h)
        lut2[h * 2048 + i] = rel_bias[bucket * H_ + h];
}

// ---------------------------------------------------------------------------
// Tiled bf16 WMMA GEMM:  C[m,n] = sum_k A[m,k] * W[n,k]   (M=4096, N=K=1024)
// Block tile 256(M) x 64(N), K-step 32, 8 waves, each wave owns a 32x64 strip
// (2 A-frags x 4 B-frags = 8 WMMAs per K-step per wave).
// AMODE: 0 = A is f32 (convert to bf16), 1 = A is bf16.
// SMODE: 0 = scatter bf16 to [b,h,s,dk]; 1 = scatter bf16 to [b,h,dk,s];
//        2 = f32 row-major [m,n].
// ---------------------------------------------------------------------------
template <int AMODE, int SMODE>
__global__ __launch_bounds__(256) void t5_gemm_kernel(const void* __restrict__ Aptr,
                                                      const float* __restrict__ W,
                                                      void* __restrict__ Cptr) {
    constexpr int K_  = 1024;
    constexpr int LDA = 40;   // bf16 elements per LDS row (32 + 8 pad)
    __shared__ bf16_t As[256 * LDA];
    __shared__ bf16_t Bs[64 * LDA];

    const int tid  = threadIdx.x;
    const int wave = tid >> 5;
    const int lane = tid & 31;
    const int n0   = blockIdx.x * 64;
    const int m0   = blockIdx.y * 256;

    v8f acc[2][4] = {{v8f{}, v8f{}, v8f{}, v8f{}}, {v8f{}, v8f{}, v8f{}, v8f{}}};

    for (int k0 = 0; k0 < K_; k0 += 32) {
        __syncthreads();
        // ---- stage A tile 256x32 -> bf16 LDS (each thread: one full row)
        {
            const int row = tid;
            bf16_t* dst = &As[row * LDA];
            if constexpr (AMODE == 0) {
                const float4* ap =
                    (const float4*)((const float*)Aptr + (size_t)(m0 + row) * K_ + k0);
#pragma unroll
                for (int j = 0; j < 4; ++j)
                    *(v8bf*)(dst + j * 8) = pack8(ap[2 * j], ap[2 * j + 1]);
                if (k0 + 32 < K_)
                    __builtin_prefetch((const float*)Aptr + (size_t)(m0 + row) * K_ + k0 + 32, 0, 0);
            } else {
                const v8bf* ap =
                    (const v8bf*)((const bf16_t*)Aptr + (size_t)(m0 + row) * K_ + k0);
#pragma unroll
                for (int j = 0; j < 4; ++j) *(v8bf*)(dst + j * 8) = ap[j];
                if (k0 + 32 < K_)
                    __builtin_prefetch((const bf16_t*)Aptr + (size_t)(m0 + row) * K_ + k0 + 32, 0, 0);
            }
        }
        // ---- stage B tile 64x32 (rows = n, contiguous k) -> bf16 LDS
        {
            const int row = tid >> 2;
            const int cs  = (tid & 3) * 8;
            const float4* wp = (const float4*)(W + (size_t)(n0 + row) * K_ + k0 + cs);
            *(v8bf*)&Bs[row * LDA + cs] = pack8(wp[0], wp[1]);
            if (k0 + 32 < K_)
                __builtin_prefetch(W + (size_t)(n0 + row) * K_ + k0 + 32 + cs, 0, 0);
        }
        __syncthreads();

        // ---- compute: 2 A-frags per wave, B-frags reused across both
        v16bf af0 = frag_a(&As[(wave * 32) * LDA], LDA);
        v16bf af1 = frag_a(&As[(wave * 32 + 16) * LDA], LDA);
#pragma unroll
        for (int nt = 0; nt < 4; ++nt) {
            v16bf bf = frag_b(&Bs[(nt * 16) * LDA], LDA);
            acc[0][nt] = wmma_bf16(af0, bf, acc[0][nt]);
            acc[1][nt] = wmma_bf16(af1, bf, acc[1][nt]);
        }
    }

    // ---- epilogue: C layout -> lane holds col (lane&15), rows (lane>>4)*8+i
#pragma unroll
    for (int mt = 0; mt < 2; ++mt) {
        const int mrow0 = m0 + wave * 32 + mt * 16 + (lane >> 4) * 8;
#pragma unroll
        for (int nt = 0; nt < 4; ++nt) {
            const int nn = n0 + nt * 16 + (lane & 15);
#pragma unroll
            for (int i = 0; i < 8; ++i) {
                const int m   = mrow0 + i;
                const float v = acc[mt][nt][i];
                if constexpr (SMODE == 2) {
                    ((float*)Cptr)[(size_t)m * 1024 + nn] = v;
                } else {
                    const int bb = m >> 10, ss = m & 1023;
                    const int hh = nn >> 6, dd = nn & 63;
                    if constexpr (SMODE == 0) {
                        ((bf16_t*)Cptr)[(((size_t)(bb * H_ + hh)) * S_ + ss) * DK_ + dd] = (bf16_t)v;
                    } else { // SMODE == 1: transposed V  [b,h,dk,s]
                        ((bf16_t*)Cptr)[(((size_t)(bb * H_ + hh)) * DK_ + dd) * S_ + ss] = (bf16_t)v;
                    }
                }
            }
        }
    }
}

// ---------------------------------------------------------------------------
// Flash attention: one block per (b, h, 64 queries); 4 waves x 16 queries.
// Q,K: [b,h,s,dk] bf16. VT: [b,h,dk,s] bf16. ctx out: [b,s,h*DK+d] bf16.
// T5: no 1/sqrt(d) scaling; additive relative bias staged per key-tile in LDS.
// ---------------------------------------------------------------------------
__global__ __launch_bounds__(128) void t5_attn_kernel(const bf16_t* __restrict__ Qg,
                                                      const bf16_t* __restrict__ Kg,
                                                      const bf16_t* __restrict__ Vg,
                                                      const float*  __restrict__ lut2,
                                                      bf16_t* __restrict__ ctx) {
    constexpr int LDT = 72;  // padded LDS row stride (bf16 elements)
    __shared__ bf16_t Kt[64 * LDT];
    __shared__ bf16_t Vt[64 * LDT];
    __shared__ bf16_t Pb[4][16 * LDT];
    __shared__ float  BiasS[128];   // contiguous rel-bias strip for this tile

    const int tid  = threadIdx.x;
    const int wave = tid >> 5;
    const int lane = tid & 31;
    const int b    = blockIdx.z;
    const int h    = blockIdx.y;
    const int q0   = blockIdx.x * 64;
    const size_t headBase = (size_t)(b * H_ + h);

    // Q fragments: 16 queries x DK=64 as two 16x32 A-frags, kept in registers.
    const bf16_t* Qp = Qg + (headBase * S_ + q0 + wave * 16) * DK_;
    const v16bf qf0 = frag_a(Qp,      DK_);
    const v16bf qf1 = frag_a(Qp + 32, DK_);

    v8f o[4] = {v8f{}, v8f{}, v8f{}, v8f{}};
    float rm[8], rl[8];
#pragma unroll
    for (int i = 0; i < 8; ++i) { rm[i] = -1e30f; rl[i] = 0.0f; }

    const int qrow = q0 + wave * 16 + (lane >> 4) * 8;  // first query row of this lane
    // bias strip offset for this lane: strip[t] holds (k-q) = (kb-q0) + t - 63
    const int bofs = (lane & 15) - wave * 16 - ((lane >> 4) << 3) + 63;

    for (int kb = 0; kb < S_; kb += 64) {
        __syncthreads();
        // ---- stage bias strip: 127 contiguous floats, coalesced
        BiasS[tid] = lut2[h * 2048 + (kb - q0) + tid + 960];

        // ---- stage K tile (keys x dk) and V^T tile (dk x keys)
        {
            const int row = tid >> 1;            // 0..63
            const int cs  = (tid & 1) * 32;      // 0 or 32
            const bf16_t* ks = Kg + (headBase * S_ + kb + row) * DK_ + cs;
            const bf16_t* vs = Vg + (headBase * DK_ + row) * S_ + kb + cs;
            bf16_t* kd = &Kt[row * LDT + cs];
            bf16_t* vd = &Vt[row * LDT + cs];
#ifdef T5_HAVE_ASYNC_LDS
#pragma unroll
            for (int j = 0; j < 4; ++j) {
                __builtin_amdgcn_global_load_async_to_lds_b128(
                    T5_TO_GM(ks + j * 8), T5_TO_LDS(kd + j * 8), 0, 0);
                __builtin_amdgcn_global_load_async_to_lds_b128(
                    T5_TO_GM(vs + j * 8), T5_TO_LDS(vd + j * 8), 0, 0);
            }
            t5_wait_async0();
#else
#pragma unroll
            for (int j = 0; j < 4; ++j) {
                ((v8bf*)kd)[j] = ((const v8bf*)ks)[j];
                ((v8bf*)vd)[j] = ((const v8bf*)vs)[j];
            }
#endif
            if (kb + 64 < S_) {
                __builtin_prefetch(ks + 64 * DK_, 0, 0);
                __builtin_prefetch(vs + 64, 0, 0);
            }
        }
        __syncthreads();

        // ---- scores: S = Q K^T   (16 q x 64 keys per wave)
        v8f sc[4];
#pragma unroll
        for (int nt = 0; nt < 4; ++nt) {
            v16bf kb0 = frag_b(&Kt[(nt * 16) * LDT], LDT);        // k = dk 0..31
            sc[nt] = wmma_bf16(qf0, kb0, v8f{});
            v16bf kb1 = frag_b(&Kt[(nt * 16) * LDT + 32], LDT);   // k = dk 32..63
            sc[nt] = wmma_bf16(qf1, kb1, sc[nt]);
        }

        // ---- additive relative position bias (from LDS strip)
#pragma unroll
        for (int nt = 0; nt < 4; ++nt)
#pragma unroll
            for (int i = 0; i < 8; ++i)
                sc[nt][i] += BiasS[bofs + nt * 16 - i];

        // ---- online softmax (row = query). Row r values live in one 16-lane
        // half (lanes 0-15 rows 0-7, lanes 16-31 rows 8-15); xor masks 1,2,4,8
        // stay within the half.
#pragma unroll
        for (int i = 0; i < 8; ++i) {
            float mx = fmaxf(fmaxf(sc[0][i], sc[1][i]), fmaxf(sc[2][i], sc[3][i]));
            mx = fmaxf(mx, __shfl_xor(mx, 1, 32));
            mx = fmaxf(mx, __shfl_xor(mx, 2, 32));
            mx = fmaxf(mx, __shfl_xor(mx, 4, 32));
            mx = fmaxf(mx, __shfl_xor(mx, 8, 32));
            const float mnew  = fmaxf(rm[i], mx);
            const float scale = __expf(rm[i] - mnew);
            rm[i] = mnew;
            float sum = 0.0f;
#pragma unroll
            for (int nt = 0; nt < 4; ++nt) {
                float p = __expf(sc[nt][i] - mnew);
                sc[nt][i] = p;
                sum += p;
            }
            sum += __shfl_xor(sum, 1, 32);
            sum += __shfl_xor(sum, 2, 32);
            sum += __shfl_xor(sum, 4, 32);
            sum += __shfl_xor(sum, 8, 32);
            rl[i] = rl[i] * scale + sum;
#pragma unroll
            for (int nt = 0; nt < 4; ++nt) o[nt][i] *= scale;
        }

        // ---- P (C-layout) -> per-wave LDS -> A-fragments
        bf16_t* Pw = &Pb[wave][0];
        const int prow0 = (lane >> 4) * 8;
        const int pcol  = lane & 15;
#pragma unroll
        for (int nt = 0; nt < 4; ++nt)
#pragma unroll
            for (int i = 0; i < 8; ++i)
                Pw[(prow0 + i) * LDT + nt * 16 + pcol] = (bf16_t)sc[nt][i];
        asm volatile("s_wait_dscnt 0" ::: "memory");   // own-wave LDS RAW

        const v16bf pf0 = frag_a(Pw,      LDT);
        const v16bf pf1 = frag_a(Pw + 32, LDT);

        // ---- O += P V   (B-frags from V^T tile: row = d, contiguous keys)
#pragma unroll
        for (int nt = 0; nt < 4; ++nt) {
            v16bf vb0 = frag_b(&Vt[(nt * 16) * LDT], LDT);        // keys 0..31
            o[nt] = wmma_bf16(pf0, vb0, o[nt]);
            v16bf vb1 = frag_b(&Vt[(nt * 16) * LDT + 32], LDT);   // keys 32..63
            o[nt] = wmma_bf16(pf1, vb1, o[nt]);
        }
    }

    // ---- normalize and store ctx as [b, s, h*DK + d] (bf16)
    float rinv[8];
#pragma unroll
    for (int i = 0; i < 8; ++i) rinv[i] = 1.0f / rl[i];
#pragma unroll
    for (int nt = 0; nt < 4; ++nt) {
        const int d = nt * 16 + (lane & 15);
#pragma unroll
        for (int i = 0; i < 8; ++i) {
            const int ss = qrow + i;
            ctx[((size_t)b * S_ + ss) * INNER + h * DK_ + d] = (bf16_t)(o[nt][i] * rinv[i]);
        }
    }
}

// ---------------------------------------------------------------------------
// Launch
// ---------------------------------------------------------------------------
extern "C" void kernel_launch(void* const* d_in, const int* in_sizes, int n_in,
                              void* d_out, int out_size, void* d_ws, size_t ws_size,
                              hipStream_t stream) {
    (void)in_sizes; (void)n_in; (void)out_size; (void)ws_size;
    const float* X    = (const float*)d_in[0];   // [B,S,D]
    const float* Wq   = (const float*)d_in[1];   // [INNER, D]
    const float* Wk   = (const float*)d_in[2];
    const float* Wv   = (const float*)d_in[3];
    const float* Wo   = (const float*)d_in[4];   // [D, INNER]
    const float* relb = (const float*)d_in[5];   // [32, H]
    float* out = (float*)d_out;                  // [B,S,D] f32

    char* ws = (char*)d_ws;
    bf16_t* Qb  = (bf16_t*)(ws);                          // 8 MB
    bf16_t* Kb  = (bf16_t*)(ws + (size_t)(8)  * 1048576); // 8 MB
    bf16_t* VT  = (bf16_t*)(ws + (size_t)(16) * 1048576); // 8 MB
    bf16_t* CTX = (bf16_t*)(ws + (size_t)(24) * 1048576); // 8 MB
    float*  LUT = (float*) (ws + (size_t)(32) * 1048576); // 16*2048*4 = 128 KB

    t5_bias_lut_kernel<<<dim3(8), dim3(256), 0, stream>>>(relb, LUT);

    dim3 gg(INNER / 64, M_ / 256);
    t5_gemm_kernel<0, 0><<<gg, 256, 0, stream>>>(X, Wq, Qb);
    t5_gemm_kernel<0, 0><<<gg, 256, 0, stream>>>(X, Wk, Kb);
    t5_gemm_kernel<0, 1><<<gg, 256, 0, stream>>>(X, Wv, VT);

    t5_attn_kernel<<<dim3(S_ / 64, H_, B_), 128, 0, stream>>>(Qb, Kb, VT, LUT, CTX);

    t5_gemm_kernel<1, 2><<<dim3(D_ / 64, M_ / 256), 256, 0, stream>>>(CTX, Wo, out);
}
